// DynamicHypergraphTower_49864570307265
// MI455X (gfx1250) — matrix operand order repaired and checked by reading, not compiled, not run on hardware
//
#include <hip/hip_runtime.h>
#include <hip/hip_bf16.h>
#include <cstdint>
#include <cstddef>

// Problem constants (match reference)
#define BB 8
#define SS 64
#define NENT 100
#define DD 1024
#define DH 256          // D/4
#define NAMED_ 768
#define D3 (3*DD)

typedef __attribute__((ext_vector_type(16))) __bf16 v16bf;
typedef __attribute__((ext_vector_type(8)))  float  v8f;

union Frag { uint4 u[2]; v16bf v; };

__device__ __forceinline__ unsigned short f2bf(float f) {
  unsigned int u = __float_as_uint(f);
  unsigned int r = u + 0x7FFFu + ((u >> 16) & 1u);   // round-to-nearest-even
  return (unsigned short)(r >> 16);
}

// -------- block reduction (sum, sumsq) for 256-thread blocks, wave32 --------
__device__ __forceinline__ void blk_reduce2(float& s0, float& s1) {
  __shared__ float sm0[8], sm1[8];
  const int lane = threadIdx.x & 31, w = threadIdx.x >> 5;
  #pragma unroll
  for (int o = 16; o > 0; o >>= 1) { s0 += __shfl_down(s0, o, 32); s1 += __shfl_down(s1, o, 32); }
  __syncthreads();
  if (lane == 0) { sm0[w] = s0; sm1[w] = s1; }
  __syncthreads();
  const int nw = blockDim.x >> 5;
  s0 = 0.f; s1 = 0.f;
  if ((int)threadIdx.x < nw) { s0 = sm0[threadIdx.x]; s1 = sm1[threadIdx.x]; }
  if (w == 0) {
    #pragma unroll
    for (int o = 4; o > 0; o >>= 1) { s0 += __shfl_down(s0, o, 32); s1 += __shfl_down(s1, o, 32); }
    if (lane == 0) { sm0[0] = s0; sm1[0] = s1; }
  }
  __syncthreads();
  s0 = sm0[0]; s1 = sm1[0];
}

// ---------------------------------------------------------------------------
// Generic bf16 WMMA GEMM: C[M,Nout] = A[M,K](lda) @ W[Nout,K]^T (+bias)(+=C)
// Block = 8 waves = 32x256 output tile; each wave: 16 rows x 64 cols.
// Double-buffered K loop (64 per iteration): the 4 WMMAs of one 32-K tile
// issue while the next tile's global loads are outstanding.
// Out-of-range A rows are address-clamped but NOT zeroed: WMMA output row m
// depends only on A row m, and out-of-range D rows are never stored.
// Requires Nout % 256 == 0, K % 64 == 0.
// ---------------------------------------------------------------------------
__global__ __launch_bounds__(256) void gemm_bf16(
    const unsigned short* __restrict__ A, int lda,
    const unsigned short* __restrict__ W,
    const float* __restrict__ bias,
    float* __restrict__ C,
    int M, int Nout, int K, int accumulate)
{
  const int lane = threadIdx.x & 31;
  const int wave = threadIdx.x >> 5;
  const int wm   = wave >> 2;          // 0..1
  const int wn   = wave & 3;           // 0..3
  const int half = lane >> 4;          // 0..1
  const int ll   = lane & 15;          // 0..15
  const int m0   = blockIdx.y * 32 + wm * 16;
  const int n0   = blockIdx.x * 256 + wn * 64;
  const int row  = m0 + ll;
  const bool rowok = row < M;
  const unsigned short* arow = A + (size_t)(rowok ? row : 0) * lda;

  // Per-column W base pointers (K-interleaved half offset folded in)
  const unsigned short* wbase[4];
  #pragma unroll
  for (int t = 0; t < 4; ++t)
    wbase[t] = W + (size_t)(n0 + t * 16 + ll) * K + half * 8;

  v8f acc[4];
  #pragma unroll
  for (int t = 0; t < 4; ++t)
    #pragma unroll
    for (int i = 0; i < 8; ++i) acc[t][i] = 0.f;

#define LOAD_A(dst, kb) do {                                          \
    (dst).u[0] = *(const uint4*)(arow + (kb) + half * 8);             \
    (dst).u[1] = *(const uint4*)(arow + (kb) + 16 + half * 8);        \
  } while (0)
#define LOAD_B(dst, kb) do {                                          \
    _Pragma("unroll")                                                 \
    for (int t = 0; t < 4; ++t) {                                     \
      const uint4* wp = (const uint4*)(wbase[t] + (kb));              \
      (dst)[t].u[0] = wp[0]; (dst)[t].u[1] = wp[2];                   \
    }                                                                 \
  } while (0)

  Frag a0, a1, b0[4], b1[4];
  LOAD_A(a0, 0);
  LOAD_B(b0, 0);
  for (int kb = 0; kb < K; kb += 64) {
    LOAD_A(a1, kb + 32);
    LOAD_B(b1, kb + 32);
    if (kb + 64 < K) __builtin_prefetch(arow + kb + 64, 0, 3);  // near-cache prefetch
    #pragma unroll
    for (int t = 0; t < 4; ++t)
      acc[t] = __builtin_amdgcn_wmma_f32_16x16x32_bf16(
          false, a0.v, false, b0[t].v, (short)0, acc[t], false, false);
    if (kb + 64 < K) {
      LOAD_A(a0, kb + 64);
      LOAD_B(b0, kb + 64);
    }
    #pragma unroll
    for (int t = 0; t < 4; ++t)
      acc[t] = __builtin_amdgcn_wmma_f32_16x16x32_bf16(
          false, a1.v, false, b1[t].v, (short)0, acc[t], false, false);
  }
#undef LOAD_A
#undef LOAD_B

  #pragma unroll
  for (int t = 0; t < 4; ++t) {
    const int col = n0 + t * 16 + ll;
    const float bb = bias ? bias[col] : 0.f;
    #pragma unroll
    for (int g = 0; g < 8; ++g) {
      const int r = m0 + half * 8 + g;          // C/D VGPR layout
      if (r < M) {
        size_t idx = (size_t)r * Nout + col;
        float v = acc[t][g] + bb;
        if (accumulate) v += C[idx];
        C[idx] = v;
      }
    }
  }
}

// --------------------------- elementwise kernels ---------------------------
__global__ void cast_f32_bf16(const float* __restrict__ src,
                              unsigned short* __restrict__ dst, int n) {
  int i = blockIdx.x * blockDim.x + threadIdx.x;
  if (i < n) dst[i] = f2bf(src[i]);
}

__global__ void copy_f32(const float* __restrict__ src, float* __restrict__ dst, int n) {
  int i = blockIdx.x * blockDim.x + threadIdx.x;
  if (i < n) dst[i] = src[i];
}

// h0 = (type_embed[tid] + LN(name_linear)) * mask ; write f32 + bf16
__global__ __launch_bounds__(256) void h0_kernel(
    const float* __restrict__ nametmp, const float* __restrict__ g, const float* __restrict__ bt,
    const float* __restrict__ type_embed, const int* __restrict__ tids,
    const unsigned char* __restrict__ emask,
    float* __restrict__ h, unsigned short* __restrict__ hbf)
{
  const int bn = blockIdx.x;
  const float* x = nametmp + (size_t)bn * DD;
  float vals[4], s0 = 0.f, s1 = 0.f;
  #pragma unroll
  for (int j = 0; j < 4; ++j) {
    float v = x[threadIdx.x + j * 256]; vals[j] = v; s0 += v; s1 += v * v;
  }
  blk_reduce2(s0, s1);
  const float mean = s0 / DD, var = s1 / DD - mean * mean;
  const float rstd = rsqrtf(var + 1e-5f);
  const int tid = tids[bn];
  const float mk = emask[bn] ? 1.f : 0.f;
  #pragma unroll
  for (int j = 0; j < 4; ++j) {
    const int d = threadIdx.x + j * 256;
    float v = (vals[j] - mean) * rstd * g[d] + bt[d];
    v = (type_embed[(size_t)tid * DD + d] + v) * mk;
    h[(size_t)bn * DD + d]  = v;
    hbf[(size_t)bn * DD + d] = f2bf(v);
  }
}

__global__ __launch_bounds__(128) void roles_kernel(
    const float* __restrict__ inc, const unsigned char* __restrict__ emask,
    float* __restrict__ roles, float* __restrict__ wsum, float* __restrict__ rsum, int s)
{
  const int b = blockIdx.x;
  float acc = 0.f;
  for (int n = threadIdx.x; n < NENT; n += 128) {
    float r = inc[((size_t)b * NENT + n) * SS + s] * (emask[b * NENT + n] ? 1.f : 0.f);
    roles[b * NENT + n] = r;
    acc += r;
  }
  __shared__ float sm[4];
  const int lane = threadIdx.x & 31, w = threadIdx.x >> 5;
  #pragma unroll
  for (int o = 16; o > 0; o >>= 1) acc += __shfl_down(acc, o, 32);
  if (lane == 0) sm[w] = acc;
  __syncthreads();
  if (threadIdx.x == 0) {
    float t = sm[0] + sm[1] + sm[2] + sm[3];
    rsum[b] = t;
    wsum[b] = fmaxf(t, 1.f);
  }
}

__global__ __launch_bounds__(256) void meanent_kernel(
    const float* __restrict__ h, const float* __restrict__ roles,
    const float* __restrict__ wsum,
    float* __restrict__ meanf, unsigned short* __restrict__ meanbf)
{
  const int b = blockIdx.y;
  const int d = blockIdx.x * 256 + threadIdx.x;
  float acc = 0.f;
  for (int n = 0; n < NENT; ++n) {
    float r = roles[b * NENT + n];            // uniform across block
    if (r != 0.f) acc += r * h[((size_t)b * NENT + n) * DD + d];
  }
  const float v = acc / wsum[b];
  meanf[b * DD + d]  = v;
  meanbf[b * DD + d] = f2bf(v);
}

// LN of eacc rows -> e_s (bf16 for GEMMs) and es history [B,S,D] f32
__global__ __launch_bounds__(256) void esln_kernel(
    const float* __restrict__ eacc, const float* __restrict__ g, const float* __restrict__ bt,
    float* __restrict__ es_hist, unsigned short* __restrict__ esbf, int s)
{
  const int b = blockIdx.x;
  float vals[4], s0 = 0.f, s1 = 0.f;
  #pragma unroll
  for (int j = 0; j < 4; ++j) {
    float v = eacc[(size_t)b * DD + threadIdx.x + j * 256]; vals[j] = v; s0 += v; s1 += v * v;
  }
  blk_reduce2(s0, s1);
  const float mean = s0 / DD, var = s1 / DD - mean * mean;
  const float rstd = rsqrtf(var + 1e-5f);
  #pragma unroll
  for (int j = 0; j < 4; ++j) {
    const int d = threadIdx.x + j * 256;
    float v = (vals[j] - mean) * rstd * g[d] + bt[d];
    es_hist[((size_t)b * SS + s) * DD + d] = v;
    esbf[b * DD + d] = f2bf(v);
  }
}

__global__ __launch_bounds__(128) void gate_kernel(
    const float* __restrict__ ghid, const float* __restrict__ W2,
    const float* __restrict__ b2, float* __restrict__ gate)
{
  const int b = blockIdx.x;
  float a0 = 0.f, a1 = 0.f, a2 = 0.f;
  for (int k = threadIdx.x; k < DH; k += 128) {
    float x = ghid[b * DH + k];
    float ge = 0.5f * x * (1.f + erff(x * 0.70710678118654752f));   // exact gelu
    a0 += ge * W2[0 * DH + k];
    a1 += ge * W2[1 * DH + k];
    a2 += ge * W2[2 * DH + k];
  }
  __shared__ float sm[3][4];
  const int lane = threadIdx.x & 31, w = threadIdx.x >> 5;
  #pragma unroll
  for (int o = 16; o > 0; o >>= 1) {
    a0 += __shfl_down(a0, o, 32); a1 += __shfl_down(a1, o, 32); a2 += __shfl_down(a2, o, 32);
  }
  if (lane == 0) { sm[0][w] = a0; sm[1][w] = a1; sm[2][w] = a2; }
  __syncthreads();
  if (threadIdx.x == 0) {
    float v0 = b2[0], v1 = b2[1], v2 = b2[2];
    for (int i = 0; i < 4; ++i) { v0 += sm[0][i]; v1 += sm[1][i]; v2 += sm[2][i]; }
    float mx = fmaxf(v0, fmaxf(v1, v2));
    float e0 = expf(v0 - mx), e1 = expf(v1 - mx), e2 = expf(v2 - mx);
    float inv = 1.f / (e0 + e1 + e2);
    gate[b * 3 + 0] = e0 * inv; gate[b * 3 + 1] = e1 * inv; gate[b * 3 + 2] = e2 * inv;
  }
}

// arc aggregation: agg[b,n,:] = normalize_t(inc*role*decay) @ es_hist[b,:s,:]
__global__ __launch_bounds__(256) void arc_kernel(
    const float* __restrict__ inc, const float* __restrict__ roles,
    const float* __restrict__ es_hist, unsigned short* __restrict__ aggbf, int s)
{
  const int n = blockIdx.x, b = blockIdx.y;
  __shared__ float w[SS];
  __shared__ float snorm;
  const float rn = roles[b * NENT + n];
  const float denom = fmaxf((float)(s - 1), 1.f);
  if ((int)threadIdx.x < SS) {
    float val = 0.f;
    if ((int)threadIdx.x < s)
      val = inc[((size_t)b * NENT + n) * SS + threadIdx.x] * rn *
            expf(-1.f + (float)threadIdx.x / denom);
    w[threadIdx.x] = val;
  }
  __syncthreads();
  if (threadIdx.x == 0) {
    float sm = 0.f;
    for (int t = 0; t < s; ++t) sm += w[t];
    snorm = 1.f / fmaxf(sm, 1e-8f);
  }
  __syncthreads();
  if (rn == 0.f) {                       // uniform: all weights zero
    #pragma unroll
    for (int j = 0; j < 4; ++j)
      aggbf[((size_t)(b * NENT + n)) * DD + threadIdx.x + j * 256] = 0;
    return;
  }
  #pragma unroll
  for (int j = 0; j < 4; ++j) {
    const int d = threadIdx.x + j * 256;
    float acc = 0.f;
    for (int t = 0; t < s; ++t) acc += w[t] * es_hist[((size_t)b * SS + t) * DD + d];
    aggbf[((size_t)(b * NENT + n)) * DD + d] = f2bf(acc * snorm);
  }
}

// interaction stream, collapsed algebraically (no NxN matmul needed):
// agg_int[n,d] = r_n * (S_d - r_n*h[n,d]) / max(r_n*(R-r_n), 1e-8),  S_d = mean*wsum
__global__ __launch_bounds__(256) void intagg_kernel(
    const float* __restrict__ h, const float* __restrict__ roles,
    const float* __restrict__ meanf, const float* __restrict__ wsum,
    const float* __restrict__ rsum, unsigned short* __restrict__ aggbf)
{
  const int b = blockIdx.y;
  const int d = blockIdx.x * 256 + threadIdx.x;
  const float sw = meanf[b * DD + d] * wsum[b];
  const float R  = rsum[b];
  for (int n = 0; n < NENT; ++n) {
    const float rn = roles[b * NENT + n];
    float out = 0.f;
    if (rn != 0.f) {
      const float dn = fmaxf(rn * (R - rn), 1e-8f);
      out = rn * (sw - rn * h[((size_t)b * NENT + n) * DD + d]) / dn;
    }
    aggbf[((size_t)(b * NENT + n)) * DD + d] = f2bf(out);
  }
}

__global__ __launch_bounds__(256) void combine_kernel(
    const float* __restrict__ scene_msg, const float* __restrict__ C1,
    const float* __restrict__ C2, const float* __restrict__ gate,
    const float* __restrict__ roles, const float* __restrict__ role_w,
    unsigned short* __restrict__ msgbf, int use_arc)
{
  const int bn = blockIdx.y;
  const int b  = bn / NENT;
  const int d  = blockIdx.x * 256 + threadIdx.x;
  const float g0 = gate[b * 3 + 0], g1 = gate[b * 3 + 1], g2 = gate[b * 3 + 2];
  const size_t idx = (size_t)bn * DD + d;
  float v = g0 * scene_msg[b * DD + d] + g2 * C2[idx];
  if (use_arc) v += g1 * C1[idx];
  v += roles[bn] * role_w[d];            // role term (zero when role==0)
  msgbf[idx] = f2bf(v);
}

// torch-semantics GRUCell + LN + clip, masked update; writes h f32 + bf16
__global__ __launch_bounds__(256) void gru_kernel(
    const float* __restrict__ gi, const float* __restrict__ gh,
    const float* __restrict__ roles, const float* __restrict__ g, const float* __restrict__ bt,
    float* __restrict__ h, unsigned short* __restrict__ hbf)
{
  const int bn = blockIdx.x;
  if (roles[bn] <= 0.f) return;          // uniform per block: h unchanged
  float pre[4], s0 = 0.f, s1 = 0.f;
  const size_t base = (size_t)bn * D3;
  #pragma unroll
  for (int j = 0; j < 4; ++j) {
    const int d = threadIdx.x + j * 256;
    const float ir = gi[base + d],          hr = gh[base + d];
    const float iz = gi[base + DD + d],     hz = gh[base + DD + d];
    const float in_ = gi[base + 2 * DD + d], hn = gh[base + 2 * DD + d];
    const float r = 1.f / (1.f + expf(-(ir + hr)));
    const float z = 1.f / (1.f + expf(-(iz + hz)));
    const float nn = tanhf(in_ + r * hn);
    const float hv = h[(size_t)bn * DD + d];
    const float p = (1.f - z) * nn + z * hv;
    pre[j] = p; s0 += p; s1 += p * p;
  }
  blk_reduce2(s0, s1);
  const float mean = s0 / DD, var = s1 / DD - mean * mean;
  const float rstd = rsqrtf(var + 1e-5f);
  #pragma unroll
  for (int j = 0; j < 4; ++j) {
    const int d = threadIdx.x + j * 256;
    float v = (pre[j] - mean) * rstd * g[d] + bt[d];
    v = fminf(50.f, fmaxf(-50.f, v));
    h[(size_t)bn * DD + d]  = v;
    hbf[(size_t)bn * DD + d] = f2bf(v);
  }
}

// H = LN(es + scene_reps) -> bf16 rows for the eo GEMM
__global__ __launch_bounds__(256) void resln_kernel(
    const float* __restrict__ es_hist, const float* __restrict__ scene,
    const float* __restrict__ g, const float* __restrict__ bt,
    unsigned short* __restrict__ outbf)
{
  const int r = blockIdx.x;              // b*S + s
  float vals[4], s0 = 0.f, s1 = 0.f;
  #pragma unroll
  for (int j = 0; j < 4; ++j) {
    const int d = threadIdx.x + j * 256;
    float v = es_hist[(size_t)r * DD + d] + scene[(size_t)r * DD + d];
    vals[j] = v; s0 += v; s1 += v * v;
  }
  blk_reduce2(s0, s1);
  const float mean = s0 / DD, var = s1 / DD - mean * mean;
  const float rstd = rsqrtf(var + 1e-5f);
  #pragma unroll
  for (int j = 0; j < 4; ++j) {
    const int d = threadIdx.x + j * 256;
    outbf[(size_t)r * DD + d] = f2bf((vals[j] - mean) * rstd * g[d] + bt[d]);
  }
}

// out = LN(gelu(X)) with eo_ln
__global__ __launch_bounds__(256) void geluln_kernel(
    const float* __restrict__ X, const float* __restrict__ g, const float* __restrict__ bt,
    float* __restrict__ out)
{
  const int r = blockIdx.x;
  float vals[4], s0 = 0.f, s1 = 0.f;
  #pragma unroll
  for (int j = 0; j < 4; ++j) {
    const int d = threadIdx.x + j * 256;
    float x = X[(size_t)r * DD + d];
    float v = 0.5f * x * (1.f + erff(x * 0.70710678118654752f));
    vals[j] = v; s0 += v; s1 += v * v;
  }
  blk_reduce2(s0, s1);
  const float mean = s0 / DD, var = s1 / DD - mean * mean;
  const float rstd = rsqrtf(var + 1e-5f);
  #pragma unroll
  for (int j = 0; j < 4; ++j) {
    const int d = threadIdx.x + j * 256;
    out[(size_t)r * DD + d] = (vals[j] - mean) * rstd * g[d] + bt[d];
  }
}

// ---------------------------------------------------------------------------
extern "C" void kernel_launch(void* const* d_in, const int* in_sizes, int n_in,
                              void* d_out, int out_size, void* d_ws, size_t ws_size,
                              hipStream_t stream) {
  (void)in_sizes; (void)n_in; (void)out_size; (void)ws_size;
  const float* scene   = (const float*)d_in[0];
  const float* inc     = (const float*)d_in[1];
  /* d_in[2] edge_type_ids: unused by reference forward */
  const int*   tids    = (const int*)d_in[3];
  const unsigned char* emask = (const unsigned char*)d_in[4];   // jnp bool = 1 byte
  const float* namee   = (const float*)d_in[5];
  const float* type_embed = (const float*)d_in[6];
  const float* name_W  = (const float*)d_in[7];
  const float* name_ln_g=(const float*)d_in[8];  const float* name_ln_b=(const float*)d_in[9];
  const float* n2e_W   = (const float*)d_in[10]; const float* n2e_b  = (const float*)d_in[11];
  const float* t2e_W   = (const float*)d_in[12]; const float* t2e_b  = (const float*)d_in[13];
  const float* edge_ln_g=(const float*)d_in[14]; const float* edge_ln_b=(const float*)d_in[15];
  const float* gate_W1 = (const float*)d_in[16]; const float* gate_b1= (const float*)d_in[17];
  const float* gate_W2 = (const float*)d_in[18]; const float* gate_b2= (const float*)d_in[19];
  const float* ms_W    = (const float*)d_in[20]; const float* ms_b   = (const float*)d_in[21];
  const float* ma_W    = (const float*)d_in[22]; const float* ma_b   = (const float*)d_in[23];
  const float* mi_W    = (const float*)d_in[24]; const float* mi_b   = (const float*)d_in[25];
  const float* role_w  = (const float*)d_in[26];
  const float* gru_Wih = (const float*)d_in[27]; const float* gru_Whh= (const float*)d_in[28];
  const float* gru_bih = (const float*)d_in[29]; const float* gru_bhh= (const float*)d_in[30];
  const float* upd_ln_g=(const float*)d_in[31];  const float* upd_ln_b=(const float*)d_in[32];
  const float* eo_W    = (const float*)d_in[33]; const float* eo_b   = (const float*)d_in[34];
  const float* eo_ln_g = (const float*)d_in[35]; const float* eo_ln_b= (const float*)d_in[36];
  const float* res_ln_g= (const float*)d_in[37]; const float* res_ln_b=(const float*)d_in[38];

  // ---- workspace bump allocator (256B aligned) ----
  uintptr_t cur = (uintptr_t)d_ws;
  auto alloc = [&](size_t bytes) -> void* {
    uintptr_t r = (cur + 255) & ~(uintptr_t)255;
    cur = r + bytes;
    return (void*)r;
  };
  const int BN = BB * NENT;        // 800
  const int BSr = BB * SS;         // 512

  unsigned short* name_Wb  = (unsigned short*)alloc((size_t)DD*NAMED_*2);
  unsigned short* n2e_Wb   = (unsigned short*)alloc((size_t)DD*DD*2);
  unsigned short* t2e_Wb   = (unsigned short*)alloc((size_t)DD*DD*2);
  unsigned short* ms_Wb    = (unsigned short*)alloc((size_t)DD*DD*2);
  unsigned short* ma_Wb    = (unsigned short*)alloc((size_t)DD*DD*2);
  unsigned short* mi_Wb    = (unsigned short*)alloc((size_t)DD*DD*2);
  unsigned short* gate_W1b = (unsigned short*)alloc((size_t)DH*DD*2);
  unsigned short* gru_Wihb = (unsigned short*)alloc((size_t)D3*DD*2);
  unsigned short* gru_Whhb = (unsigned short*)alloc((size_t)D3*DD*2);
  unsigned short* eo_Wb    = (unsigned short*)alloc((size_t)DD*DD*2);
  unsigned short* scene_bf = (unsigned short*)alloc((size_t)BSr*DD*2);
  unsigned short* name_bf  = (unsigned short*)alloc((size_t)BN*NAMED_*2);
  unsigned short* hbf      = (unsigned short*)alloc((size_t)BN*DD*2);
  unsigned short* esbf     = (unsigned short*)alloc((size_t)BB*DD*2);
  unsigned short* meanbf   = (unsigned short*)alloc((size_t)BB*DD*2);
  unsigned short* agg_arc_bf = (unsigned short*)alloc((size_t)BN*DD*2);
  unsigned short* agg_int_bf = (unsigned short*)alloc((size_t)BN*DD*2);
  unsigned short* msgbf    = (unsigned short*)alloc((size_t)BN*DD*2);
  unsigned short* Hres_bf  = (unsigned short*)alloc((size_t)BSr*DD*2);
  float* nametmp  = (float*)alloc((size_t)BN*DD*4);
  float* h        = (float*)alloc((size_t)BN*DD*4);
  float* es_hist  = (float*)alloc((size_t)BSr*DD*4);
  float* eacc     = (float*)alloc((size_t)BB*DD*4);
  float* meanf    = (float*)alloc((size_t)BB*DD*4);
  float* scene_msg= (float*)alloc((size_t)BB*DD*4);
  float* gatehid  = (float*)alloc((size_t)BB*DH*4);
  float* gate     = (float*)alloc((size_t)BB*3*4);
  float* roles    = (float*)alloc((size_t)BN*4);
  float* wsum     = (float*)alloc((size_t)BB*4);
  float* rsum     = (float*)alloc((size_t)BB*4);
  float* C1       = (float*)alloc((size_t)BN*DD*4);
  float* C2       = (float*)alloc((size_t)BN*DD*4);
  float* gi       = (float*)alloc((size_t)BN*D3*4);
  float* gh       = (float*)alloc((size_t)BN*D3*4);
  float* eo_tmp   = (float*)alloc((size_t)BSr*DD*4);

  auto castk = [&](const float* src, unsigned short* dst, int n) {
    cast_f32_bf16<<<(n + 255) / 256, 256, 0, stream>>>(src, dst, n);
  };
  auto gemm = [&](const unsigned short* A, int lda, const unsigned short* W,
                  const float* bias, float* C, int M, int Nout, int K, int accum) {
    dim3 grid(Nout / 256, (M + 31) / 32);
    gemm_bf16<<<grid, 256, 0, stream>>>(A, lda, W, bias, C, M, Nout, K, accum);
  };

  // ---- setup: bf16 weight/activation copies ----
  castk(name_W,  name_Wb,  DD * NAMED_);
  castk(n2e_W,   n2e_Wb,   DD * DD);
  castk(t2e_W,   t2e_Wb,   DD * DD);
  castk(ms_W,    ms_Wb,    DD * DD);
  castk(ma_W,    ma_Wb,    DD * DD);
  castk(mi_W,    mi_Wb,    DD * DD);
  castk(gate_W1, gate_W1b, DH * DD);
  castk(gru_Wih, gru_Wihb, D3 * DD);
  castk(gru_Whh, gru_Whhb, D3 * DD);
  castk(eo_W,    eo_Wb,    DD * DD);
  castk(scene,   scene_bf, BSr * DD);
  castk(namee,   name_bf,  BN * NAMED_);

  // h0 = type_embed[tid] + LN(name_embs @ name_W^T), masked
  gemm(name_bf, NAMED_, name_Wb, nullptr, nametmp, BN, DD, NAMED_, 0);
  h0_kernel<<<BN, 256, 0, stream>>>(nametmp, name_ln_g, name_ln_b, type_embed,
                                    tids, emask, h, hbf);

  // ---- sequential scan over S steps ----
  for (int s = 0; s < SS; ++s) {
    roles_kernel<<<BB, 128, 0, stream>>>(inc, emask, roles, wsum, rsum, s);
    meanent_kernel<<<dim3(DD / 256, BB), 256, 0, stream>>>(h, roles, wsum, meanf, meanbf);
    gemm(meanbf, DD, n2e_Wb, n2e_b, eacc, BB, DD, DD, 0);
    gemm(scene_bf + (size_t)s * DD, SS * DD, t2e_Wb, t2e_b, eacc, BB, DD, DD, 1);
    esln_kernel<<<BB, 256, 0, stream>>>(eacc, edge_ln_g, edge_ln_b, es_hist, esbf, s);
    gemm(esbf, DD, ms_Wb, ms_b, scene_msg, BB, DD, DD, 0);
    gemm(esbf, DD, gate_W1b, gate_b1, gatehid, BB, DH, DD, 0);
    gate_kernel<<<BB, 128, 0, stream>>>(gatehid, gate_W2, gate_b2, gate);
    if (s > 0) {
      arc_kernel<<<dim3(NENT, BB), 256, 0, stream>>>(inc, roles, es_hist, agg_arc_bf, s);
      gemm(agg_arc_bf, DD, ma_Wb, ma_b, C1, BN, DD, DD, 0);
    }
    intagg_kernel<<<dim3(DD / 256, BB), 256, 0, stream>>>(h, roles, meanf, wsum, rsum, agg_int_bf);
    gemm(agg_int_bf, DD, mi_Wb, mi_b, C2, BN, DD, DD, 0);
    combine_kernel<<<dim3(DD / 256, BN), 256, 0, stream>>>(
        scene_msg, C1, C2, gate, roles, role_w, msgbf, (s > 0) ? 1 : 0);
    gemm(msgbf, DD, gru_Wihb, gru_bih, gi, BN, D3, DD, 0);
    gemm(hbf,   DD, gru_Whhb, gru_bhh, gh, BN, D3, DD, 0);
    gru_kernel<<<BN, 256, 0, stream>>>(gi, gh, roles, upd_ln_g, upd_ln_b, h, hbf);
  }

  // ---- output head ----
  resln_kernel<<<BSr, 256, 0, stream>>>(es_hist, scene, res_ln_g, res_ln_b, Hres_bf);
  gemm(Hres_bf, DD, eo_Wb, eo_b, eo_tmp, BSr, DD, DD, 0);
  geluln_kernel<<<BSr, 256, 0, stream>>>(eo_tmp, eo_ln_g, eo_ln_b, (float*)d_out);
  copy_f32<<<(BN * DD + 255) / 256, 256, 0, stream>>>(h, (float*)d_out + (size_t)BSr * DD,
                                                      BN * DD);
}
